// MMFIModel_25872882991135
// MI455X (gfx1250) — matrix operand order repaired and checked by reading, not compiled, not run on hardware
//
#include <hip/hip_runtime.h>

// ---------------------------------------------------------------------------
// MMFI forward for MI455X (gfx1250).
//   - bf16 WMMA (v_wmma_f32_16x16x32_bf16) for every matmul
//   - weight tiles staged into LDS via TDM tensor_load_to_lds (fallback: coop copy)
//   - 2 M-tiles per wave sharing one B fragment
//   - f32 VALU kernels for pointwise / softmax / batch-norm passes
// ---------------------------------------------------------------------------

typedef __bf16 bf16_t;
typedef __attribute__((ext_vector_type(8)))  __bf16 bf16x8;
typedef __attribute__((ext_vector_type(16))) __bf16 bf16x16;
typedef __attribute__((ext_vector_type(8)))  float  f32x8;
typedef __attribute__((ext_vector_type(4)))  unsigned int uint32x4;
typedef __attribute__((ext_vector_type(8)))  int int32x8;
typedef __attribute__((ext_vector_type(4)))  int int32x4;

#ifndef __has_builtin
#define __has_builtin(x) 0
#endif
#if defined(__HIP_DEVICE_COMPILE__) && \
    __has_builtin(__builtin_amdgcn_tensor_load_to_lds) && \
    __has_builtin(__builtin_amdgcn_s_wait_tensorcnt)
#define USE_TDM 1
#else
#define USE_TDM 0
#endif

#define DEV __device__ __forceinline__

static __device__ const float kEps = 1e-5f;

// ---- WMMA A fragment loader (16x16x32 bf16, wave32) ------------------------
// A (MxK) 16x32 tile: lanes 0-15 row m=lane, K = {k0..k0+7, k0+16..k0+23};
//                     lanes16-31 row m=lane-16, K = {k0+8..k0+15, k0+24..k0+31}
DEV bf16x16 load_a_frag(const bf16_t* A, int lda, int row_base, int k0, int lane) {
  const int half = lane >> 4;
  const int m    = lane & 15;
  const bf16_t* p = A + (long long)(row_base + m) * lda + k0 + half * 8;
  bf16x8 lo = *(const bf16x8*)p;
  bf16x8 hi = *(const bf16x8*)(p + 16);
  return __builtin_shufflevector(lo, hi, 0,1,2,3,4,5,6,7,8,9,10,11,12,13,14,15);
}

// ---- Generic batched GEMM:  C[z] = A[z](MxK,bf16) * Bw[z](NxK,bf16)^T + bias
// Weight layout requirement: ldb == K (all weight buffers are padded that way).
// Each block stages its 16xK weight tile in LDS (TDM when available); each wave
// computes two 16x16 output tiles sharing one B fragment.
__global__ void k_gemm_bf16(const bf16_t* __restrict__ A, const bf16_t* __restrict__ Bw,
                            const float* __restrict__ bias, float* __restrict__ C,
                            int M, int N, int K, int lda, int ldc,
                            long long sA, long long sB, long long sC, long long sBias)
{
  __shared__ bf16_t sBt[16 * 256];        // max K = 256 -> 8KB
  const int lane  = threadIdx.x & 31;
  const int wave  = threadIdx.x >> 5;
  const int wpb   = blockDim.x >> 5;
  const int ntile = blockIdx.y;
  const int z     = blockIdx.z;
  const bf16_t* Bz = Bw + (long long)z * sB + (long long)ntile * 16 * K;

  // ---- stage B tile (16 rows x K, contiguous since ldb==K) into LDS ----
#if USE_TDM
  if (wave == 0) {
    const unsigned lds_off = (unsigned)(uintptr_t)(&sBt[0]);
    const unsigned long long ga = (unsigned long long)(uintptr_t)Bz;
    const unsigned elems = 16u * (unsigned)K;           // <= 4096, fits tile_dim0
    uint32x4 g0;
    g0.x = 1u;                                          // count = 1 valid descriptor
    g0.y = lds_off;                                     // lds_addr
    g0.z = (unsigned)(ga & 0xFFFFFFFFu);                // global_addr[31:0]
    g0.w = (unsigned)((ga >> 32) & 0x1FFFFFFu) | (2u << 30); // addr[56:32] | type=2
    int32x8 g1;
    g1[0] = (int)(1u << 16);                            // data_size = 2 bytes
    g1[1] = (int)((elems & 0xFFFFu) << 16);             // tensor_dim0[15:0]
    g1[2] = (int)(((elems >> 16) & 0xFFFFu) | (1u << 16)); // dim0 hi | tensor_dim1 = 1
    g1[3] = (int)((elems & 0xFFFFu) << 16);             // tile_dim0 = elems
    g1[4] = 0;                                          // tile_dim1/2 unused
    g1[5] = (int)elems;                                 // tensor_dim0_stride lo
    g1[6] = 0;
    g1[7] = 0;
    int32x4 g2z = {0, 0, 0, 0};
#if __clang_major__ >= 23
    int32x8 g4z = {0, 0, 0, 0, 0, 0, 0, 0};
    __builtin_amdgcn_tensor_load_to_lds(g0, g1, g2z, g2z, g4z, 0);
#else
    __builtin_amdgcn_tensor_load_to_lds(g0, g1, g2z, g2z, 0);
#endif
    __builtin_amdgcn_s_wait_tensorcnt((short)0);
  }
#else
  {
    const int total = 16 * K;                           // multiple of 8
    for (int i = threadIdx.x * 8; i < total; i += blockDim.x * 8)
      *(bf16x8*)&sBt[i] = *(const bf16x8*)&Bz[i];
  }
#endif
  __syncthreads();

  const int mtile = blockIdx.x * wpb + wave;
  const int row0  = mtile * 32;
  if (row0 >= M) return;                 // grids are exact; never taken
  const bf16_t* Az = A + (long long)z * sA;
  float*        Cz = C + (long long)z * sC;

  f32x8 acc0 = {0.f,0.f,0.f,0.f,0.f,0.f,0.f,0.f};
  f32x8 acc1 = {0.f,0.f,0.f,0.f,0.f,0.f,0.f,0.f};
  const int half = lane >> 4;
  const int n    = lane & 15;
  for (int k0 = 0; k0 < K; k0 += 32) {
    if (k0 + 32 < K)  // hint next K-slice of A into cache (global_prefetch_b8)
      __builtin_prefetch((const void*)(Az + (long long)(row0 + (lane & 15)) * lda + k0 + 32), 0, 0);
    bf16x16 a0 = load_a_frag(Az, lda, row0,      k0, lane);
    bf16x16 a1 = load_a_frag(Az, lda, row0 + 16, k0, lane);
    // B frag from LDS: lanes 0-15 col n, K={k0..k0+7, k0+16..k0+23}; lanes 16-31 col n, K={k0+8..k0+15, k0+24..k0+31}
    bf16x8 blo = *(const bf16x8*)&sBt[n * K + k0 + half * 8];
    bf16x8 bhi = *(const bf16x8*)&sBt[n * K + k0 + 16 + half * 8];
    bf16x16 b  = __builtin_shufflevector(blo, bhi, 0,1,2,3,4,5,6,7,8,9,10,11,12,13,14,15);
    acc0 = __builtin_amdgcn_wmma_f32_16x16x32_bf16(false, a0, false, b, (short)0, acc0, false, false);
    acc1 = __builtin_amdgcn_wmma_f32_16x16x32_bf16(false, a1, false, b, (short)0, acc1, false, false);
  }
  const int col = ntile * 16 + n;
  if (col < N) {
    const float bv = bias ? bias[(long long)z * sBias + col] : 0.f;
    const int rb = row0 + half * 8;
#pragma unroll
    for (int v = 0; v < 8; ++v)
      Cz[(long long)(rb + v) * ldc + col] = acc0[v] + bv;
#pragma unroll
    for (int v = 0; v < 8; ++v)
      Cz[(long long)(rb + 16 + v) * ldc + col] = acc1[v] + bv;
  }
}

// ---- weight convert + zero-pad:  dst(z,dR,dC) bf16  <-  src(z,rows,cols) f32
__global__ void k_cvt_pad(const float* __restrict__ src, bf16_t* __restrict__ dst,
                          int batch, int rows, int cols,
                          long long srcBatchStride, int srcRowStride, int srcColOff,
                          int dstRows, int dstCols)
{
  long long tid = (long long)blockIdx.x * blockDim.x + threadIdx.x;
  long long total = (long long)batch * dstRows * dstCols;
  if (tid >= total) return;
  int c = (int)(tid % dstCols);
  long long r1 = tid / dstCols;
  int r = (int)(r1 % dstRows);
  int z = (int)(r1 / dstRows);
  float v = 0.f;
  if (r < rows && c < cols)
    v = src[(long long)z * srcBatchStride + (long long)r * srcRowStride + srcColOff + c];
  dst[tid] = (bf16_t)v;
}

// ---- zero the K-pad columns of a bf16 activation matrix --------------------
__global__ void k_padzero(bf16_t* __restrict__ p, long long rows, int ld, int start)
{
  const int padw = ld - start;
  long long tid = (long long)blockIdx.x * blockDim.x + threadIdx.x;
  long long total = rows * padw;
  if (tid >= total) return;
  long long r = tid / padw;
  int c = start + (int)(tid % padw);
  p[r * ld + c] = (bf16_t)0.f;
}

// ---- fields / lfields / sti ------------------------------------------------
__global__ void k_fields(const int* __restrict__ cat, const float* __restrict__ num,
                         const float* __restrict__ emb, const float* __restrict__ lin,
                         const float* __restrict__ uW, const float* __restrict__ ub,
                         const float* __restrict__ iW, const float* __restrict__ ib,
                         const float* __restrict__ ulW, const float* __restrict__ ulb,
                         const float* __restrict__ ilW, const float* __restrict__ ilb,
                         const float* __restrict__ scen_emb, const float* __restrict__ task_emb,
                         const float* __restrict__ st_w,
                         float* __restrict__ fields, float* __restrict__ lfields,
                         bf16_t* __restrict__ lfields_bf, float* __restrict__ sti)
{
  const int b = blockIdx.x;
  const int e = threadIdx.x;              // 128
  const int* cb = cat + b * 10;
  for (int j = 0; j < 9; ++j) {
    const int row = cb[1 + j] + j * 1000;
    const float fv = emb[(long long)row * 128 + e];
    const float lv = lin[(long long)row * 128 + e];
    fields [((long long)b * 11 + j) * 128 + e] = fv;
    lfields[((long long)b * 11 + j) * 128 + e] = lv;
    lfields_bf[((long long)b * 11 + j) * 128 + e] = (bf16_t)lv;
  }
  const float* nb = num + b * 63;
  float u = ub[e];
  for (int i = 0; i < 23; ++i) u += nb[i] * uW[e * 23 + i];
  float it = ib[e];
  for (int i = 0; i < 40; ++i) it += nb[23 + i] * iW[e * 40 + i];
  float ul = ulb[e];
  for (int i = 0; i < 23; ++i) ul += nb[i] * ulW[e * 23 + i];
  float il = ilb[e];
  for (int i = 0; i < 40; ++i) il += nb[23 + i] * ilW[e * 40 + i];
  fields [((long long)b * 11 + 9 ) * 128 + e] = u;
  fields [((long long)b * 11 + 10) * 128 + e] = it;
  lfields[((long long)b * 11 + 9 ) * 128 + e] = ul;
  lfields[((long long)b * 11 + 10) * 128 + e] = il;
  lfields_bf[((long long)b * 11 + 9 ) * 128 + e] = (bf16_t)ul;
  lfields_bf[((long long)b * 11 + 10) * 128 + e] = (bf16_t)il;
  if (e < 8) {                            // sti[b][t][si]
    const int t = e >> 2, si = e & 3;
    const int sc = cb[0];
    sti[((long long)b * 2 + t) * 4 + si] =
        scen_emb[sc * 4 + si] * task_emb[t * 4 + si] * st_w[t * 4 + sc];
  }
}

// ---- pairwise interactions pe(b,f,e) -> bf16 -------------------------------
DEV void f2rc(int f, int& r, int& c) {
  int rr = 0, rem = f, cnt = 10;
  while (rem >= cnt) { rem -= cnt; ++rr; --cnt; }
  r = rr; c = rr + 1 + rem;
}
__global__ void k_pe(const float* __restrict__ fields, const float* __restrict__ interact_w,
                     bf16_t* __restrict__ pe)
{
  const long long bf = blockIdx.x;
  const int e = threadIdx.x;
  const int f = (int)(bf % 55);
  const long long b = bf / 55;
  int r, c; f2rc(f, r, c);
  const float v = fields[(b * 11 + r) * 128 + e] * fields[(b * 11 + c) * 128 + e] * interact_w[f];
  pe[bf * 128 + e] = (bf16_t)v;
}

// ---- MoE combine: experts + gate softmax -> w (bf16, K-padded) -------------
__global__ void k_moe(const float* __restrict__ ein, const float* __restrict__ glog,
                      const float* __restrict__ sti,
                      const float* __restrict__ shW, const float* __restrict__ shb,
                      const float* __restrict__ spW, const float* __restrict__ spb,
                      const float* __restrict__ gW,
                      bf16_t* __restrict__ w, int F, int ldw, int Bn)
{
  long long tid = (long long)blockIdx.x * blockDim.x + threadIdx.x;
  long long total = (long long)8 * Bn * F;
  if (tid >= total) return;
  const int f = (int)(tid % F);
  long long r1 = tid / F;
  const int b = (int)(r1 % Bn);
  const int st = (int)(r1 / Bn);      // st = s*T + t
  const int t = st & 1;

  float e_in[14];
  const float* ep = ein + ((long long)b * F + f) * 14;
#pragma unroll
  for (int i = 0; i < 14; ++i) e_in[i] = ep[i];

  float ex[6][4];
#pragma unroll
  for (int g = 0; g < 4; ++g)
#pragma unroll
    for (int o = 0; o < 4; ++o) {
      const float* wp = shW + (g * 4 + o) * 14;
      float a = shb[g * 4 + o];
      for (int i = 0; i < 14; ++i) a += e_in[i] * wp[i];
      ex[g][o] = a;
    }
#pragma unroll
  for (int p = 0; p < 2; ++p)
#pragma unroll
    for (int o = 0; o < 4; ++o) {
      const float* wp = spW + ((st * 2 + p) * 4 + o) * 14;
      float a = spb[(st * 2 + p) * 4 + o];
      for (int i = 0; i < 14; ++i) a += e_in[i] * wp[i];
      ex[4 + p][o] = a;
    }

  float gl[6];
  const float* stib = sti + ((long long)b * 2 + t) * 4;
#pragma unroll
  for (int g = 0; g < 6; ++g) {
    const float* g4 = gW + (long long)(st * 6 + g) * 132;
    float a = glog[((long long)b * F + f) * 48 + st * 6 + g];
    for (int i = 0; i < 4; ++i) a += stib[i] * g4[i];
    gl[g] = a;
  }
  float mx = gl[0];
  for (int g = 1; g < 6; ++g) mx = fmaxf(mx, gl[g]);
  float sm = 0.f;
#pragma unroll
  for (int g = 0; g < 6; ++g) { gl[g] = __expf(gl[g] - mx); sm += gl[g]; }
  const float inv = 1.f / sm;

  bf16_t* wp2 = w + ((long long)st * Bn + b) * ldw + f * 4;
#pragma unroll
  for (int o = 0; o < 4; ++o) {
    float a = 0.f;
#pragma unroll
    for (int g = 0; g < 6; ++g) a += ex[g][o] * gl[g];
    wp2[o] = (bf16_t)(a * inv);
  }
}

// ---- batch-norm statistics over axis b -------------------------------------
__global__ void k_stats(const float* __restrict__ X, float* __restrict__ mean_out,
                        float* __restrict__ rsig_out, int Bn, int C, int ld)
{
  const int zc = blockIdx.x;
  const int c = zc % C;
  const int z = zc / C;
  const float* Xz = X + (long long)z * Bn * ld;
  float s = 0.f, s2 = 0.f;
  for (int b = threadIdx.x; b < Bn; b += blockDim.x) {
    const float v = Xz[(long long)b * ld + c];
    s += v; s2 += v * v;
  }
  __shared__ float sh1[256], sh2[256];
  sh1[threadIdx.x] = s; sh2[threadIdx.x] = s2;
  __syncthreads();
  for (int o = blockDim.x >> 1; o > 0; o >>= 1) {
    if (threadIdx.x < (unsigned)o) {
      sh1[threadIdx.x] += sh1[threadIdx.x + o];
      sh2[threadIdx.x] += sh2[threadIdx.x + o];
    }
    __syncthreads();
  }
  if (threadIdx.x == 0) {
    const float m = sh1[0] / Bn;
    const float var = sh2[0] / Bn - m * m;
    mean_out[zc] = m;
    rsig_out[zc] = rsqrtf(var + kEps);
  }
}

// ---- attention readout -> h[:, :128] (bf16) --------------------------------
__global__ void k_attout(const int* __restrict__ cat, const float* __restrict__ m_pre,
                         const float* __restrict__ m_mean, const float* __restrict__ m_rsig,
                         const float* __restrict__ att_W, const bf16_t* __restrict__ pe,
                         bf16_t* __restrict__ h, int Bn)
{
  const int tb = blockIdx.x;
  const int e = threadIdx.x;            // 128
  const int b = tb % Bn;
  const int t = tb / Bn;
  const int s = cat[b * 10];
  const int st = s * 2 + t;
  __shared__ float coef[55];
  for (int f = threadIdx.x; f < 55; f += blockDim.x) {
    float v = m_pre[((long long)st * Bn + b) * 55 + f];
    v = (v - m_mean[st * 55 + f]) * m_rsig[st * 55 + f];
    v = tanhf(v);
    v = fmaxf(v, 0.f);
    coef[f] = v * att_W[st * 55 + f];
  }
  __syncthreads();
  float acc = 0.f;
  const bf16_t* peb = pe + (long long)b * 55 * 128 + e;
  for (int f = 0; f < 55; ++f) acc += coef[f] * (float)peb[f * 128];
  h[((long long)t * Bn + b) * 256 + e] = (bf16_t)acc;
}

// ---- linear-path readout -> h[:, 128:256] (bf16) ---------------------------
__global__ void k_linfinal(const int* __restrict__ cat, const float* __restrict__ lt_pre,
                           const float* __restrict__ lt_mean, const float* __restrict__ lt_rsig,
                           const float* __restrict__ lfields, const float* __restrict__ bias_p,
                           bf16_t* __restrict__ h, int Bn)
{
  const int tb = blockIdx.x;
  const int e = threadIdx.x;            // 128
  const int b = tb % Bn;
  const int t = tb / Bn;
  const int s = cat[b * 10];
  const int st = s * 2 + t;
  __shared__ float lw[11];
  if (threadIdx.x < 11) {
    const int f = threadIdx.x;
    float v = lt_pre[((long long)st * Bn + b) * 11 + f];
    lw[f] = (v - lt_mean[st * 11 + f]) * lt_rsig[st * 11 + f];
  }
  __syncthreads();
  if (threadIdx.x == 0) {
    float mx = lw[0];
    for (int f = 1; f < 11; ++f) mx = fmaxf(mx, lw[f]);
    float sm = 0.f;
    for (int f = 0; f < 11; ++f) { lw[f] = __expf(lw[f] - mx); sm += lw[f]; }
    const float inv = 1.f / sm;
    for (int f = 0; f < 11; ++f) lw[f] *= inv;
  }
  __syncthreads();
  float acc = bias_p[((long long)t * 4 + s) * 128 + e];
  const float* lf = lfields + (long long)b * 11 * 128 + e;
  for (int f = 0; f < 11; ++f) acc += lw[f] * lf[f * 128];
  h[((long long)t * Bn + b) * 256 + 128 + e] = (bf16_t)acc;
}

// ---- BN + ReLU -> bf16 (K-padded) ------------------------------------------
__global__ void k_bnrelu(const float* __restrict__ X, const float* __restrict__ mean,
                         const float* __restrict__ rsig, bf16_t* __restrict__ Y,
                         long long rows, int C, int Cpad, int Bn)
{
  long long tid = (long long)blockIdx.x * blockDim.x + threadIdx.x;
  long long total = rows * Cpad;
  if (tid >= total) return;
  long long r = tid / Cpad;
  int c = (int)(tid % Cpad);
  int z = (int)(r / Bn);
  float v = 0.f;
  if (c < C) {
    v = X[r * C + c];
    v = (v - mean[z * C + c]) * rsig[z * C + c];
    v = fmaxf(v, 0.f);
  }
  Y[r * Cpad + c] = (bf16_t)v;
}

// ---- final 8->1 + sigmoid --------------------------------------------------
__global__ void k_final(const float* __restrict__ h3, const float* __restrict__ mean,
                        const float* __restrict__ rsig, const float* __restrict__ W4,
                        const float* __restrict__ b4, float* __restrict__ out, int Bn)
{
  long long tid = (long long)blockIdx.x * blockDim.x + threadIdx.x;
  if (tid >= (long long)2 * Bn) return;
  const int b = (int)(tid % Bn);
  const int t = (int)(tid / Bn);
  float acc = b4[t];
#pragma unroll
  for (int i = 0; i < 8; ++i) {
    float v = h3[((long long)t * Bn + b) * 8 + i];
    v = (v - mean[t * 8 + i]) * rsig[t * 8 + i];
    v = fmaxf(v, 0.f);
    acc += v * W4[t * 8 + i];
  }
  out[(long long)t * Bn + b] = 1.f / (1.f + __expf(-acc));
}

// ===========================================================================
extern "C" void kernel_launch(void* const* d_in, const int* in_sizes, int n_in,
                              void* d_out, int out_size, void* d_ws, size_t ws_size,
                              hipStream_t stream)
{
  (void)n_in; (void)out_size; (void)ws_size;
  const int Bn = in_sizes[0] / 10;      // 4096

  const int*   cat        = (const int*)  d_in[0];
  const float* num        = (const float*)d_in[1];
  const float* emb        = (const float*)d_in[2];
  const float* lin        = (const float*)d_in[3];
  const float* uW         = (const float*)d_in[4];
  const float* ubv        = (const float*)d_in[5];
  const float* iW         = (const float*)d_in[6];
  const float* ibv        = (const float*)d_in[7];
  const float* ulW        = (const float*)d_in[8];
  const float* ulb        = (const float*)d_in[9];
  const float* ilW        = (const float*)d_in[10];
  const float* ilb        = (const float*)d_in[11];
  const float* scen_emb   = (const float*)d_in[12];
  const float* task_emb   = (const float*)d_in[13];
  const float* st_w       = (const float*)d_in[14];
  const float* interact_w = (const float*)d_in[15];
  const float* sb_W       = (const float*)d_in[16];
  const float* sb_b       = (const float*)d_in[17];
  const float* lsb_W      = (const float*)d_in[18];
  const float* lsb_b      = (const float*)d_in[19];
  const float* shexp_W    = (const float*)d_in[20];
  const float* shexp_b    = (const float*)d_in[21];
  const float* spexp_W    = (const float*)d_in[22];
  const float* spexp_b    = (const float*)d_in[23];
  const float* lshexp_W   = (const float*)d_in[24];
  const float* lshexp_b   = (const float*)d_in[25];
  const float* lspexp_W   = (const float*)d_in[26];
  const float* lspexp_b   = (const float*)d_in[27];
  const float* gate_W     = (const float*)d_in[28];
  const float* gate_b     = (const float*)d_in[29];
  const float* lgate_W    = (const float*)d_in[30];
  const float* lgate_b    = (const float*)d_in[31];
  const float* mask_W     = (const float*)d_in[32];
  const float* mask_b     = (const float*)d_in[33];
  const float* att_W      = (const float*)d_in[34];
  const float* ltower_W   = (const float*)d_in[35];
  const float* ltower_b   = (const float*)d_in[36];
  const float* bias_p     = (const float*)d_in[37];
  const float* W1         = (const float*)d_in[38];
  const float* b1         = (const float*)d_in[39];
  const float* W2         = (const float*)d_in[40];
  const float* b2         = (const float*)d_in[41];
  const float* W3         = (const float*)d_in[42];
  const float* b3         = (const float*)d_in[43];
  const float* W4         = (const float*)d_in[44];
  const float* b4         = (const float*)d_in[45];
  float* out = (float*)d_out;

  char* wsb = (char*)d_ws;
  size_t off = 0;
  auto alloc = [&](size_t bytes) -> void* {
    void* p = wsb + off;
    off += (bytes + 255) & ~(size_t)255;
    return p;
  };

  // activations / intermediates
  float*  fields     = (float*) alloc((size_t)Bn * 11 * 128 * 4);
  float*  lfields    = (float*) alloc((size_t)Bn * 11 * 128 * 4);
  bf16_t* lfields_bf = (bf16_t*)alloc((size_t)Bn * 11 * 128 * 2);
  float*  sti        = (float*) alloc((size_t)Bn * 2 * 4 * 4);
  bf16_t* pe_bf      = (bf16_t*)alloc((size_t)Bn * 55 * 128 * 2);
  float*  einA       = (float*) alloc((size_t)Bn * 55 * 14 * 4);
  float*  einL       = (float*) alloc((size_t)Bn * 11 * 14 * 4);
  float*  glog       = (float*) alloc((size_t)Bn * 55 * 48 * 4);
  float*  glogL      = (float*) alloc((size_t)Bn * 11 * 48 * 4);
  bf16_t* w_bf       = (bf16_t*)alloc((size_t)8 * Bn * 224 * 2);
  bf16_t* w2_bf      = (bf16_t*)alloc((size_t)8 * Bn * 64 * 2);
  float*  m_pre      = (float*) alloc((size_t)8 * Bn * 55 * 4);
  float*  lt_pre     = (float*) alloc((size_t)8 * Bn * 11 * 4);
  float*  m_mean     = (float*) alloc(8 * 55 * 4);
  float*  m_rsig     = (float*) alloc(8 * 55 * 4);
  float*  lt_mean    = (float*) alloc(8 * 11 * 4);
  float*  lt_rsig    = (float*) alloc(8 * 11 * 4);
  bf16_t* h_bf       = (bf16_t*)alloc((size_t)2 * Bn * 256 * 2);
  float*  h1_pre     = (float*) alloc((size_t)2 * Bn * 40 * 4);
  bf16_t* h1_bf      = (bf16_t*)alloc((size_t)2 * Bn * 64 * 2);
  float*  h2_pre     = (float*) alloc((size_t)2 * Bn * 32 * 4);
  bf16_t* h2_bf      = (bf16_t*)alloc((size_t)2 * Bn * 32 * 2);
  float*  h3_pre     = (float*) alloc((size_t)2 * Bn * 8 * 4);
  float*  h1_mean    = (float*) alloc(2 * 40 * 4);
  float*  h1_rsig    = (float*) alloc(2 * 40 * 4);
  float*  h2_mean    = (float*) alloc(2 * 32 * 4);
  float*  h2_rsig    = (float*) alloc(2 * 32 * 4);
  float*  h3_mean    = (float*) alloc(2 * 8 * 4);
  float*  h3_rsig    = (float*) alloc(2 * 8 * 4);
  // bf16 padded weights (all stored with ldb == K of their GEMM)
  bf16_t* sbW_bf  = (bf16_t*)alloc((size_t)16 * 128 * 2);
  bf16_t* lsbW_bf = (bf16_t*)alloc((size_t)16 * 128 * 2);
  bf16_t* gwE_bf  = (bf16_t*)alloc((size_t)48 * 128 * 2);
  bf16_t* lgwE_bf = (bf16_t*)alloc((size_t)48 * 128 * 2);
  bf16_t* mkW_bf  = (bf16_t*)alloc((size_t)8 * 64 * 224 * 2);
  bf16_t* ltW_bf  = (bf16_t*)alloc((size_t)8 * 16 * 64 * 2);
  bf16_t* W1_bf   = (bf16_t*)alloc((size_t)2 * 48 * 256 * 2);
  bf16_t* W2_bf   = (bf16_t*)alloc((size_t)2 * 32 * 64 * 2);
  bf16_t* W3_bf   = (bf16_t*)alloc((size_t)2 * 16 * 32 * 2);

  const int TPB = 256;
  auto cdiv = [](long long a, long long b) -> unsigned { return (unsigned)((a + b - 1) / b); };

  // ---- weight prep ----
  k_cvt_pad<<<cdiv((long long)16*128, TPB), TPB, 0, stream>>>(sb_W,  sbW_bf,  1, 14, 128, 0, 128, 0, 16, 128);
  k_cvt_pad<<<cdiv((long long)16*128, TPB), TPB, 0, stream>>>(lsb_W, lsbW_bf, 1, 14, 128, 0, 128, 0, 16, 128);
  k_cvt_pad<<<cdiv((long long)48*128, TPB), TPB, 0, stream>>>(gate_W,  gwE_bf,  1, 48, 128, 0, 132, 4, 48, 128);
  k_cvt_pad<<<cdiv((long long)48*128, TPB), TPB, 0, stream>>>(lgate_W, lgwE_bf, 1, 48, 128, 0, 132, 4, 48, 128);
  k_cvt_pad<<<cdiv((long long)8*64*224, TPB), TPB, 0, stream>>>(mask_W,   mkW_bf, 8, 55, 220, 55*220, 220, 0, 64, 224);
  k_cvt_pad<<<cdiv((long long)8*16*64,  TPB), TPB, 0, stream>>>(ltower_W, ltW_bf, 8, 11, 44,  11*44,  44,  0, 16, 64);
  k_cvt_pad<<<cdiv((long long)2*48*256, TPB), TPB, 0, stream>>>(W1, W1_bf, 2, 40, 256, 40*256, 256, 0, 48, 256);
  k_cvt_pad<<<cdiv((long long)2*32*64,  TPB), TPB, 0, stream>>>(W2, W2_bf, 2, 32, 40,  32*40,  40,  0, 32, 64);
  k_cvt_pad<<<cdiv((long long)2*16*32,  TPB), TPB, 0, stream>>>(W3, W3_bf, 2, 8,  32,  8*32,   32,  0, 16, 32);

  // ---- fields + pe ----
  k_fields<<<Bn, 128, 0, stream>>>(cat, num, emb, lin, uW, ubv, iW, ibv,
                                   ulW, ulb, ilW, ilb, scen_emb, task_emb, st_w,
                                   fields, lfields, lfields_bf, sti);
  k_pe<<<Bn * 55, 128, 0, stream>>>(fields, interact_w, pe_bf);

  // ---- ein / gate-logit GEMMs (WMMA, 32 rows per wave) ----
  const int M55 = Bn * 55, M11 = Bn * 11;
  k_gemm_bf16<<<dim3(cdiv(M55/32, 8), 1, 1), TPB, 0, stream>>>(
      pe_bf, sbW_bf, sb_b, einA, M55, 14, 128, 128, 14, 0, 0, 0, 0);
  k_gemm_bf16<<<dim3(cdiv(M55/32, 8), 3, 1), TPB, 0, stream>>>(
      pe_bf, gwE_bf, gate_b, glog, M55, 48, 128, 128, 48, 0, 0, 0, 0);
  k_gemm_bf16<<<dim3(cdiv(M11/32, 8), 1, 1), TPB, 0, stream>>>(
      lfields_bf, lsbW_bf, lsb_b, einL, M11, 14, 128, 128, 14, 0, 0, 0, 0);
  k_gemm_bf16<<<dim3(cdiv(M11/32, 8), 3, 1), TPB, 0, stream>>>(
      lfields_bf, lgwE_bf, lgate_b, glogL, M11, 48, 128, 128, 48, 0, 0, 0, 0);

  // ---- MoE combine -> w (bf16, padded K) ----
  k_moe<<<cdiv((long long)8*Bn*55, TPB), TPB, 0, stream>>>(
      einA, glog, sti, shexp_W, shexp_b, spexp_W, spexp_b, gate_W, w_bf, 55, 224, Bn);
  k_moe<<<cdiv((long long)8*Bn*11, TPB), TPB, 0, stream>>>(
      einL, glogL, sti, lshexp_W, lshexp_b, lspexp_W, lspexp_b, lgate_W, w2_bf, 11, 64, Bn);
  k_padzero<<<cdiv((long long)8*Bn*(224-220), TPB), TPB, 0, stream>>>(w_bf,  (long long)8*Bn, 224, 220);
  k_padzero<<<cdiv((long long)8*Bn*(64-44),   TPB), TPB, 0, stream>>>(w2_bf, (long long)8*Bn, 64,  44);

  // ---- mask / ltower batched GEMMs ----
  k_gemm_bf16<<<dim3(cdiv(Bn/32, 8), 4, 8), TPB, 0, stream>>>(
      w_bf, mkW_bf, mask_b, m_pre, Bn, 55, 224, 224, 55,
      (long long)Bn*224, 64*224, (long long)Bn*55, 55);
  k_gemm_bf16<<<dim3(cdiv(Bn/32, 8), 1, 8), TPB, 0, stream>>>(
      w2_bf, ltW_bf, ltower_b, lt_pre, Bn, 11, 64, 64, 11,
      (long long)Bn*64, 16*64, (long long)Bn*11, 11);

  // ---- BN stats + readouts ----
  k_stats<<<8*55, TPB, 0, stream>>>(m_pre,  m_mean,  m_rsig,  Bn, 55, 55);
  k_stats<<<8*11, TPB, 0, stream>>>(lt_pre, lt_mean, lt_rsig, Bn, 11, 11);
  k_attout  <<<2*Bn, 128, 0, stream>>>(cat, m_pre, m_mean, m_rsig, att_W, pe_bf, h_bf, Bn);
  k_linfinal<<<2*Bn, 128, 0, stream>>>(cat, lt_pre, lt_mean, lt_rsig, lfields, bias_p, h_bf, Bn);

  // ---- head: 256 -> 40 -> 32 -> 8 -> 1 (BN+ReLU between) ----
  k_gemm_bf16<<<dim3(cdiv(Bn/32, 8), 3, 2), TPB, 0, stream>>>(
      h_bf, W1_bf, b1, h1_pre, Bn, 40, 256, 256, 40,
      (long long)Bn*256, 48*256, (long long)Bn*40, 40);
  k_stats<<<2*40, TPB, 0, stream>>>(h1_pre, h1_mean, h1_rsig, Bn, 40, 40);
  k_bnrelu<<<cdiv((long long)2*Bn*64, TPB), TPB, 0, stream>>>(h1_pre, h1_mean, h1_rsig, h1_bf,
                                                              (long long)2*Bn, 40, 64, Bn);
  k_gemm_bf16<<<dim3(cdiv(Bn/32, 8), 2, 2), TPB, 0, stream>>>(
      h1_bf, W2_bf, b2, h2_pre, Bn, 32, 64, 64, 32,
      (long long)Bn*64, 32*64, (long long)Bn*32, 32);
  k_stats<<<2*32, TPB, 0, stream>>>(h2_pre, h2_mean, h2_rsig, Bn, 32, 32);
  k_bnrelu<<<cdiv((long long)2*Bn*32, TPB), TPB, 0, stream>>>(h2_pre, h2_mean, h2_rsig, h2_bf,
                                                              (long long)2*Bn, 32, 32, Bn);
  k_gemm_bf16<<<dim3(cdiv(Bn/32, 8), 1, 2), TPB, 0, stream>>>(
      h2_bf, W3_bf, b3, h3_pre, Bn, 8, 32, 32, 8,
      (long long)Bn*32, 16*32, (long long)Bn*8, 8);
  k_stats<<<2*8, TPB, 0, stream>>>(h3_pre, h3_mean, h3_rsig, Bn, 8, 8);
  k_final<<<cdiv((long long)2*Bn, TPB), TPB, 0, stream>>>(h3_pre, h3_mean, h3_rsig, W4, b4, out, Bn);
}
